// MultiHeadedAttention_5033701671177
// MI455X (gfx1250) — matrix operand order repaired and checked by reading, not compiled, not run on hardware
//
#include <hip/hip_runtime.h>
#include <hip/hip_bf16.h>

typedef __bf16 bf16;
typedef __attribute__((ext_vector_type(16))) __bf16 v16bf;
typedef __attribute__((ext_vector_type(8)))  __bf16 v8bf;
typedef __attribute__((ext_vector_type(8)))  float  v8f;
typedef __attribute__((ext_vector_type(4)))  float  v4f;

#define NH   16
#define DM   1024
#define DK   64
#define SEQ  2048
#define MTOT 4096   // B*S

__device__ __forceinline__ v16bf cat16(v8bf lo, v8bf hi) {
  v16bf r;
#pragma unroll
  for (int i = 0; i < 8; ++i) { r[i] = lo[i]; r[i + 8] = hi[i]; }
  return r;
}

// A-operand fragment from bf16 row: lane<16 holds K {kb..kb+7, kb+16..kb+23}, kb=(lane>>4)*8
__device__ __forceinline__ v16bf frag_a(const bf16* rowp, int kb) {
  v8bf lo = *(const v8bf*)(rowp + kb);
  v8bf hi = *(const v8bf*)(rowp + kb + 16);
  return cat16(lo, hi);
}
// B-operand fragment from bf16 row (we feed B^T rows): lane holds K {ks..ks+7, ks+16..ks+23}, ks=(lane>>4)*8
__device__ __forceinline__ v16bf frag_b(const bf16* rowp, int ks) {
  v8bf lo = *(const v8bf*)(rowp + ks);
  v8bf hi = *(const v8bf*)(rowp + ks + 16);
  return cat16(lo, hi);
}
// Same fragments but sourced from fp32 LDS (async-staged), converting on the fly
__device__ __forceinline__ v16bf frag_a_f32(const float* rowp, int kb) {
  const float* p = rowp + kb;
  v16bf r;
#pragma unroll
  for (int e = 0; e < 8; ++e) r[e] = (bf16)p[e];
#pragma unroll
  for (int e = 0; e < 8; ++e) r[e + 8] = (bf16)p[16 + e];
  return r;
}
__device__ __forceinline__ v16bf frag_b_f32(const float* rowp, int ks) {
  const float* p = rowp + ks;
  v16bf r;
#pragma unroll
  for (int e = 0; e < 16; ++e) r[e] = (bf16)p[(e & 7) + ((e >> 3) << 4)];
  return r;
}

__device__ __forceinline__ v8f wmma_bf16(v16bf a, v16bf b, v8f c) {
  return __builtin_amdgcn_wmma_f32_16x16x32_bf16(false, a, false, b, (short)0, c, false, false);
}

// async copy 16B global -> LDS (ASYNCcnt path); ldsByteAddr = wave-relative LDS byte offset
__device__ __forceinline__ void async_g2l_b128(unsigned ldsByteAddr, const float* gptr) {
  asm volatile("global_load_async_to_lds_b128 %0, %1, off"
               :: "v"(ldsByteAddr), "v"(gptr) : "memory");
}
__device__ __forceinline__ void wait_async0() {
  asm volatile("s_wait_asynccnt 0x0" ::: "memory");
}

// ---------------------------------------------------------------------------
// Y = X[M x 1024] * W[1024 x 1024]^T + bias   (torch nn.Linear convention)
// Double-buffered LDS staging via async global->LDS copies (copy i+1 overlaps WMMA i).
// MODE 0: fp32 row-major [M,1024] (final output projection)
// MODE 1: bf16 head-split  out[bh][s][dk]   (Q, K)
// MODE 2: bf16 head-transposed out[bh][dk][s]  (V)
// ---------------------------------------------------------------------------
template <int MODE>
__global__ __launch_bounds__(256) void proj_gemm(const float* __restrict__ X,
                                                 const float* __restrict__ W,
                                                 const float* __restrict__ bias,
                                                 void* __restrict__ out) {
  constexpr int BM = 128, BN = 64, BK = 32, LDA = 36;  // fp32 LDS, 144B row stride (16B aligned)
  constexpr int NK = DM / BK;
  __shared__ float Asf[2][BM * LDA];
  __shared__ float Bsf[2][BN * LDA];

  const int tid  = threadIdx.x;
  const int lane = tid & 31;
  const int wave = tid >> 5;
  const int wm   = wave & 3;          // 4 waves along M
  const int wn   = wave >> 2;         // 2 waves along N
  const int blockM = blockIdx.x * BM;
  const int blockN = blockIdx.y * BN;
  const int lrow = lane & 15;
  const int kb   = (lane >> 4) << 3;  // A frag K base: 0 / 8
  const int ks   = (lane >> 4) << 3;  // B frag K base: 0 / 8

  // fixed per-thread staging slots (16B granules), per LDS buffer
  unsigned aDst[2][4]; const float* aSrc[4];
  unsigned bDst[2][2]; const float* bSrc[2];
#pragma unroll
  for (int r = 0; r < 4; ++r) {
    int idx = tid + r * 256;               // 0..1023 : 128 rows x 8 granules
    int row = idx >> 3, c4 = (idx & 7) << 2;
    aDst[0][r] = (unsigned)(size_t)(&Asf[0][0] + row * LDA + c4);
    aDst[1][r] = (unsigned)(size_t)(&Asf[1][0] + row * LDA + c4);
    aSrc[r]    = X + (size_t)(blockM + row) * DM + c4;
  }
#pragma unroll
  for (int r = 0; r < 2; ++r) {
    int idx = tid + r * 256;               // 0..511 : 64 rows x 8 granules
    int row = idx >> 3, c4 = (idx & 7) << 2;
    bDst[0][r] = (unsigned)(size_t)(&Bsf[0][0] + row * LDA + c4);
    bDst[1][r] = (unsigned)(size_t)(&Bsf[1][0] + row * LDA + c4);
    bSrc[r]    = W + (size_t)(blockN + row) * DM + c4;
  }

  v8f acc[2][2] = {};

  // prologue: stage k-tile 0 into buffer 0
#pragma unroll
  for (int r = 0; r < 4; ++r) async_g2l_b128(aDst[0][r], aSrc[r]);
#pragma unroll
  for (int r = 0; r < 2; ++r) async_g2l_b128(bDst[0][r], bSrc[r]);

  for (int i = 0; i < NK; ++i) {
    const int cur = i & 1, nxt = cur ^ 1;
    wait_async0();        // our copies into `cur` have landed
    __syncthreads();      // everyone's copies into `cur` landed; everyone done reading `nxt`

    if (i + 1 < NK) {     // overlap: stage k-tile i+1 while computing k-tile i
      const int kk = (i + 1) * BK;
#pragma unroll
      for (int r = 0; r < 4; ++r) async_g2l_b128(aDst[nxt][r], aSrc[r] + kk);
#pragma unroll
      for (int r = 0; r < 2; ++r) async_g2l_b128(bDst[nxt][r], bSrc[r] + kk);
    }

    v16bf af[2], bf_[2];
#pragma unroll
    for (int mt = 0; mt < 2; ++mt)
      af[mt] = frag_a_f32(&Asf[cur][0] + (wm * 32 + mt * 16 + lrow) * LDA, kb);
#pragma unroll
    for (int nt = 0; nt < 2; ++nt)
      bf_[nt] = frag_b_f32(&Bsf[cur][0] + (wn * 32 + nt * 16 + lrow) * LDA, ks);

#pragma unroll
    for (int mt = 0; mt < 2; ++mt)
#pragma unroll
      for (int nt = 0; nt < 2; ++nt)
        acc[mt][nt] = wmma_bf16(af[mt], bf_[nt], acc[mt][nt]);
  }

  // epilogue: C/D layout -> VGPR r holds (M = r + 8*(lane>=16), N = lane&15)
  const int mbase = blockM + wm * 32;
  const int nbase = blockN + wn * 32;
  const int i0 = (lane >> 4) << 3;
#pragma unroll
  for (int mt = 0; mt < 2; ++mt) {
#pragma unroll
    for (int nt = 0; nt < 2; ++nt) {
      const int gn = nbase + nt * 16 + lrow;
      const float bv = bias[gn];
      const int m0 = mbase + mt * 16 + i0;
      v8f c = acc[mt][nt];
      if (MODE == 0) {
        float* o = (float*)out;
#pragma unroll
        for (int r = 0; r < 8; ++r)
          o[(size_t)(m0 + r) * DM + gn] = c[r] + bv;
      } else if (MODE == 1) {
        bf16* o = (bf16*)out;
        const int b = m0 >> 11;                 // /SEQ (tile never crosses batch)
        const int h = gn >> 6, d = gn & 63;
#pragma unroll
        for (int r = 0; r < 8; ++r) {
          int s = (m0 + r) & (SEQ - 1);
          o[(((size_t)(b * NH + h)) * SEQ + s) * DK + d] = (bf16)(c[r] + bv);
        }
      } else {  // MODE 2: transposed [bh][d][s], 8 contiguous s -> one b128 store
        bf16* o = (bf16*)out;
        const int b = m0 >> 11;
        const int h = gn >> 6, d = gn & 63;
        const int s0 = m0 & (SEQ - 1);
        v8bf hv;
#pragma unroll
        for (int r = 0; r < 8; ++r) hv[r] = (bf16)(c[r] + bv);
        *(v8bf*)(o + (((size_t)(b * NH + h)) * DK + d) * SEQ + s0) = hv;
      }
    }
  }
}

// ---------------------------------------------------------------------------
// Attention: one block per (bh, 16-query tile). scores -> softmax -> P*V
// Q,K bf16 [bh][s][dk]; V bf16 transposed [bh][dk][s]; ctx fp32 [B,S,H*DK]
// ---------------------------------------------------------------------------
#define SLD 2052  // padded fp32 row stride (2052 % 64 == 4 -> no bank conflicts; 8208B, 16B aligned)

__global__ __launch_bounds__(256) void attn_kernel(const bf16* __restrict__ Qb,
                                                   const bf16* __restrict__ Kb,
                                                   const bf16* __restrict__ Vt,
                                                   float* __restrict__ ctx) {
  __shared__ float sc[16 * SLD];      // ~128 KB score rows
  __shared__ float ctxAcc[16 * DK];   // 4 KB

  const int tid  = threadIdx.x;
  const int lane = tid & 31;
  const int wave = tid >> 5;
  const int bh = blockIdx.x;   // 0..31
  const int qt = blockIdx.y;   // 0..127
  const int lrow = lane & 15;
  const int kb = (lane >> 4) << 3;
  const int ks = (lane >> 4) << 3;
  const int i0 = (lane >> 4) << 3;

  for (int i = tid; i < 16 * DK; i += 256) ctxAcc[i] = 0.f;

  // Q A-operand fragments: 16x64 -> two K=32 steps (reused across all key tiles)
  const bf16* qp = Qb + ((size_t)bh * SEQ + qt * 16 + lrow) * DK;
  v16bf aq[2];
#pragma unroll
  for (int s = 0; s < 2; ++s) aq[s] = frag_a(qp + s * 32, kb);

  // ---- scores = (Q K^T) / 8 : wave handles key tiles jt = wave + 8*t ----
  for (int t = 0; t < 16; ++t) {
    const int jt = wave + t * 8;
    const bf16* kp = Kb + ((size_t)bh * SEQ + jt * 16 + lrow) * DK;
    if (t + 1 < 16)
      __builtin_prefetch(Kb + ((size_t)bh * SEQ + (jt + 8) * 16 + lrow) * DK, 0, 1);
    v8f s8 = {};
#pragma unroll
    for (int s = 0; s < 2; ++s)
      s8 = wmma_bf16(aq[s], frag_b(kp + s * 32, ks), s8);
    const int j = jt * 16 + lrow;
#pragma unroll
    for (int r = 0; r < 8; ++r) sc[(i0 + r) * SLD + j] = s8[r] * 0.125f;
  }
  __syncthreads();

  // ---- softmax: each wave owns rows 2*wave, 2*wave+1 ----
#pragma unroll
  for (int rr = 0; rr < 2; ++rr) {
    float* p = sc + (wave * 2 + rr) * SLD;
    float mx = -3.0e38f;
    for (int c = lane; c < SEQ; c += 32) mx = fmaxf(mx, p[c]);
#pragma unroll
    for (int off = 16; off; off >>= 1) mx = fmaxf(mx, __shfl_xor(mx, off, 32));
    float sum = 0.f;
    for (int c = lane; c < SEQ; c += 32) {
      float e = __expf(p[c] - mx);
      p[c] = e;
      sum += e;
    }
#pragma unroll
    for (int off = 16; off; off >>= 1) sum += __shfl_xor(sum, off, 32);
    const float inv = 1.f / sum;
    for (int c = lane; c < SEQ; c += 32) p[c] *= inv;
  }
  __syncthreads();

  // ---- ctx = P * V : wave owns key range [wave*256, wave*256+256) ----
  v8f o[4] = {};
  const int j0 = wave * 256;
  for (int kstep = 0; kstep < 8; ++kstep) {
    const int kt = j0 + kstep * 32;
    // A-frag of P: fp32 LDS -> bf16
    const float* pr = sc + lrow * SLD + kt + kb;
    v16bf ap;
#pragma unroll
    for (int e = 0; e < 8; ++e) ap[e] = (bf16)pr[e];
#pragma unroll
    for (int e = 0; e < 8; ++e) ap[e + 8] = (bf16)pr[16 + e];
#pragma unroll
    for (int nt = 0; nt < 4; ++nt) {
      const bf16* vp = Vt + ((size_t)bh * DK + nt * 16 + lrow) * SEQ + kt;
      o[nt] = wmma_bf16(ap, frag_b(vp, ks), o[nt]);
    }
  }
#pragma unroll
  for (int nt = 0; nt < 4; ++nt) {
    const int d = nt * 16 + lrow;
#pragma unroll
    for (int r = 0; r < 8; ++r) atomicAdd(&ctxAcc[(i0 + r) * DK + d], o[nt][r]);
  }
  __syncthreads();

  // write ctx [B, S, H*DK]
  const int b = bh >> 4, h = bh & 15;
  for (int i = tid; i < 16 * DK; i += 256) {
    const int r = i >> 6, d = i & 63;
    const int s = qt * 16 + r;
    ctx[((size_t)b * SEQ + s) * DM + h * DK + d] = ctxAcc[i];
  }
}

// ---------------------------------------------------------------------------
extern "C" void kernel_launch(void* const* d_in, const int* in_sizes, int n_in,
                              void* d_out, int out_size, void* d_ws, size_t ws_size,
                              hipStream_t stream) {
  (void)in_sizes; (void)n_in; (void)out_size; (void)ws_size;
  const float* q  = (const float*)d_in[0];
  const float* k  = (const float*)d_in[1];
  const float* v  = (const float*)d_in[2];
  const float* Wq = (const float*)d_in[3];
  const float* bq = (const float*)d_in[4];
  const float* Wk = (const float*)d_in[5];
  const float* bk = (const float*)d_in[6];
  const float* Wv = (const float*)d_in[7];
  const float* bv = (const float*)d_in[8];
  const float* Wo = (const float*)d_in[9];
  const float* bo = (const float*)d_in[10];

  const size_t headElems = (size_t)2 * NH * SEQ * DK;  // 4M bf16 elems per tensor
  bf16* Qb = (bf16*)d_ws;
  bf16* Kb = Qb + headElems;
  bf16* Vt = Kb + headElems;
  float* ctx = (float*)(Vt + headElems);

  dim3 blk(256);
  dim3 gp(MTOT / 128, DM / 64);  // 32 x 16
  proj_gemm<1><<<gp, blk, 0, stream>>>(q, Wq, bq, (void*)Qb);
  proj_gemm<1><<<gp, blk, 0, stream>>>(k, Wk, bk, (void*)Kb);
  proj_gemm<2><<<gp, blk, 0, stream>>>(v, Wv, bv, (void*)Vt);

  attn_kernel<<<dim3(2 * NH, SEQ / 16), blk, 0, stream>>>(Qb, Kb, Vt, ctx);

  proj_gemm<0><<<gp, blk, 0, stream>>>(ctx, Wo, bo, d_out);
}
